// UnifiedDecoder_79113297592771
// MI455X (gfx1250) — matrix operand order repaired and checked by reading, not compiled, not run on hardware
//
#include <hip/hip_runtime.h>
#include <math.h>

// Problem constants (match reference)
#define D_MODEL 384
#define N_HEAD  8
#define HDIM    48
#define T_DEC   512
#define N_ENC   1024
#define B_SZ    4
#define L_LAYERS 6
#define F_FFN   1536
#define V_VOCAB 8000

typedef __attribute__((ext_vector_type(8)))  float  v8f;
typedef __attribute__((ext_vector_type(8)))  __bf16 v8bf;
typedef __attribute__((ext_vector_type(16))) __bf16 v16bf;

__device__ __forceinline__ unsigned short f2bf(float f) {
  unsigned int u = __float_as_uint(f);
  unsigned int r = (u + 0x7fffu + ((u >> 16) & 1u)) >> 16;  // RNE
  return (unsigned short)r;
}

__device__ __forceinline__ unsigned int pack2bf(float a, float b) {
  return (unsigned int)f2bf(a) | ((unsigned int)f2bf(b) << 16);
}

// Build a 16-element bf16 fragment from two 16-byte LDS chunks.
__device__ __forceinline__ v16bf load_frag(const unsigned short* p0,
                                           const unsigned short* p1) {
  v8bf lo = *(const v8bf*)p0;
  v8bf hi = *(const v8bf*)p1;
  return __builtin_shufflevector(lo, hi, 0,1,2,3,4,5,6,7,8,9,10,11,12,13,14,15);
}

__device__ __forceinline__ unsigned int lds_addr_of(const void* p) {
  return (unsigned int)(unsigned long long)p;  // generic->LDS: low 32 bits
}

// ---------------------------------------------------------------------------
// Weight convert + transpose: in fp32 [K][N] -> out bf16 [N][K]
// ---------------------------------------------------------------------------
__global__ __launch_bounds__(256) void w2bf_kernel(
    const float* __restrict__ in, unsigned short* __restrict__ out,
    int K, int N) {
  int idx = blockIdx.x * 256 + threadIdx.x;
  if (idx >= K * N) return;
  int k = idx / N;
  int n = idx - k * N;
  out[(size_t)n * K + k] = f2bf(in[idx]);
}

// ---------------------------------------------------------------------------
// Embedding + sinusoidal positional encoding: x[b*T+t, :] = emb[id] + pe(t)
// ---------------------------------------------------------------------------
__global__ __launch_bounds__(128) void embed_kernel(
    const float* __restrict__ emb, const int* __restrict__ ids,
    float* __restrict__ x) {
  const int row = blockIdx.x;           // b*T + t
  const int t   = row & (T_DEC - 1);
  const int id  = ids[row];
  const int tid = threadIdx.x;
#pragma unroll
  for (int j = 0; j < 3; ++j) {
    int d  = tid + j * 128;
    int i2 = d >> 1;
    float ang = (float)t *
        __expf((float)(2 * i2) * (-9.210340371976184f / (float)D_MODEL));
    float pe  = (d & 1) ? __cosf(ang) : __sinf(ang);
    x[(size_t)row * D_MODEL + d] = emb[(size_t)id * D_MODEL + d] + pe;
  }
}

// ---------------------------------------------------------------------------
// GEMM: C[M,Nc] = act(A[M,K] @ W[K,Nc] + bias)
//  - A: fp32 activations, converted to bf16 while staging to LDS
//  - Wb: pre-converted bf16 weights, TRANSPOSED layout [Nc][K]
//  - W tiles staged with global_load_async_to_lds_b128 (ASYNCcnt path)
//  - Double-buffered, software-pipelined K loop: next tile's async W copy
//    and A global loads are issued before the current tile's WMMAs.
// block = 256 threads (8 waves), block tile 128x64,
// wave tile 32x32 => 4 v_wmma_f32_16x16x32_bf16 per K-step per wave.
// grid = (Nc/64, M/128). K mult of 32, Nc mult of 64, M mult of 128.
// ---------------------------------------------------------------------------
template <int ACT>
__global__ __launch_bounds__(256) void gemm_kernel(
    const float* __restrict__ A, const unsigned short* __restrict__ Wb,
    const float* __restrict__ bias, float* __restrict__ C,
    int K, int Nc) {
  __shared__ __align__(16) unsigned short As[2][128 * 40];  // [m][k] (+8 pad)
  __shared__ __align__(16) unsigned short Wt[2][64 * 32];   // [n][k]

  const int tid  = threadIdx.x;
  const int lane = tid & 31;
  const int w    = tid >> 5;
  const int wr   = w >> 1;        // 0..3 : 32-row tile within 128
  const int wc   = w & 1;         // 0..1 : 32-col tile within 64
  const int row0 = blockIdx.y * 128;
  const int col0 = blockIdx.x * 64;
  const int ln16 = lane & 15;
  const int hi   = lane >> 4;

  // A staging map: 2 threads per row, 16 floats each
  const int ami = tid >> 1;             // 0..127
  const int aki = (tid & 1) * 16;       // 0 or 16
  // W async staging map: 4 threads per n-row, 16B each
  const int wn  = tid >> 2;             // 0..63
  const int wko = (tid & 3) * 8;        // half-element offset within 32

  const float* arow = A + (size_t)(row0 + ami) * K + aki;
  const unsigned short* wrow = Wb + (size_t)(col0 + wn) * K + wko;

  const unsigned int wt_lds0 = lds_addr_of(&Wt[0][wn * 32 + wko]);
  const unsigned int wt_lds1 = lds_addr_of(&Wt[1][wn * 32 + wko]);

  v8f acc00 = {}, acc01 = {}, acc10 = {}, acc11 = {};

  // ---- prologue: stage tile k0=0 into buffer 0 ----
  {
    unsigned long long ga = (unsigned long long)(wrow);
    asm volatile("global_load_async_to_lds_b128 %0, %1, off"
                 :: "v"(wt_lds0), "v"(ga) : "memory");
#pragma unroll
    for (int j = 0; j < 4; ++j) {
      const float4 va = *(const float4*)(arow + j * 4);
      uint2 pk;
      pk.x = pack2bf(va.x, va.y);
      pk.y = pack2bf(va.z, va.w);
      *(uint2*)&As[0][ami * 40 + aki + j * 4] = pk;
    }
    asm volatile("s_wait_asynccnt 0" ::: "memory");
  }
  __syncthreads();

  int cur = 0;
  for (int k0 = 0; k0 < K; k0 += 32) {
    const int nxt = cur ^ 1;
    const bool havenext = (k0 + 32) < K;

    // --- issue next tile's loads before computing current tile ---
    float4 areg[4];
    if (havenext) {
      unsigned long long ga = (unsigned long long)(wrow + (k0 + 32));
      asm volatile("global_load_async_to_lds_b128 %0, %1, off"
                   :: "v"(nxt ? wt_lds1 : wt_lds0), "v"(ga) : "memory");
#pragma unroll
      for (int j = 0; j < 4; ++j)
        areg[j] = *(const float4*)(arow + (k0 + 32) + j * 4);
    }

    // --- fragments + 4 WMMA on current buffer ---
    const unsigned short* asb = &As[cur][0];
    const unsigned short* wsb = &Wt[cur][0];
    const unsigned short* a0p = &asb[(wr * 32 + ln16) * 40 + hi * 16];
    v16bf af0 = load_frag(a0p, a0p + 8);
    const unsigned short* a1p = &asb[(wr * 32 + 16 + ln16) * 40 + hi * 16];
    v16bf af1 = load_frag(a1p, a1p + 8);
    const unsigned short* b0p = &wsb[(wc * 32 + ln16) * 32 + hi * 16];
    v16bf bf0 = load_frag(b0p, b0p + 8);
    const unsigned short* b1p = &wsb[(wc * 32 + 16 + ln16) * 32 + hi * 16];
    v16bf bf1 = load_frag(b1p, b1p + 8);

    acc00 = __builtin_amdgcn_wmma_f32_16x16x32_bf16(
        false, af0, false, bf0, (short)0, acc00, false, false);
    acc01 = __builtin_amdgcn_wmma_f32_16x16x32_bf16(
        false, af0, false, bf1, (short)0, acc01, false, false);
    acc10 = __builtin_amdgcn_wmma_f32_16x16x32_bf16(
        false, af1, false, bf0, (short)0, acc10, false, false);
    acc11 = __builtin_amdgcn_wmma_f32_16x16x32_bf16(
        false, af1, false, bf1, (short)0, acc11, false, false);

    if (havenext) {
      // convert + store next A tile, then drain async W copy, one barrier
#pragma unroll
      for (int j = 0; j < 4; ++j) {
        uint2 pk;
        pk.x = pack2bf(areg[j].x, areg[j].y);
        pk.y = pack2bf(areg[j].z, areg[j].w);
        *(uint2*)&As[nxt][ami * 40 + aki + j * 4] = pk;
      }
      asm volatile("s_wait_asynccnt 0" ::: "memory");
      __syncthreads();
    }
    cur = nxt;
  }

  const int ocol0 = col0 + wc * 32 + ln16;
  const float bv0 = bias[ocol0];
  const float bv1 = bias[ocol0 + 16];
  const int orow0 = row0 + wr * 32 + hi * 8;

  v8f* accs[4] = { &acc00, &acc01, &acc10, &acc11 };
#pragma unroll
  for (int s = 0; s < 2; ++s) {
#pragma unroll
    for (int t2 = 0; t2 < 2; ++t2) {
      const v8f& a = *accs[s * 2 + t2];
      const int orow = orow0 + s * 16;
      const int ocol = ocol0 + t2 * 16;
      const float bv = t2 ? bv1 : bv0;
#pragma unroll
      for (int r = 0; r < 8; ++r) {
        float val = a[r] + bv;
        if (ACT == 1)  // exact GELU
          val = 0.5f * val * (1.0f + erff(val * 0.70710678118654752f));
        C[(size_t)(orow + r) * Nc + ocol] = val;
      }
    }
  }
}

// ---------------------------------------------------------------------------
// Fused flash attention, one wave32 per (b, h, 16-query tile).
// q rows: [b*T + t][D] head offset h*HDIM; k/v rows: [b*Tk + n][D].
// HD=48 zero-padded to 64 for QK^T; key tiles of 32; online softmax.
// grid = (T/16, H, B), block = 32
// ---------------------------------------------------------------------------
__global__ __launch_bounds__(32) void attn_kernel(
    const float* __restrict__ q, const float* __restrict__ k,
    const float* __restrict__ v, float* __restrict__ o,
    int Tk, int causal) {
  __shared__ __align__(16) unsigned short Qs[16 * 64];   // [m][kd]
  __shared__ __align__(16) unsigned short Ks[32 * 64];   // [key][kd]
  __shared__ __align__(16) unsigned short Vts[48 * 40];  // [kd][key] (+pad)
  __shared__ __align__(16) unsigned short Ps[16 * 40];   // [m][key]  (+pad)

  const int qt = blockIdx.x, h = blockIdx.y, b = blockIdx.z;
  const int lane = threadIdx.x;
  const int ln16 = lane & 15;
  const int hi   = lane >> 4;

  const size_t qbase = ((size_t)b * T_DEC + qt * 16) * D_MODEL + h * HDIM;
  for (int idx = lane; idx < 16 * 64; idx += 32) {
    int m = idx >> 6, kd = idx & 63;
    float val = (kd < HDIM) ? q[qbase + (size_t)m * D_MODEL + kd] : 0.0f;
    Qs[idx] = f2bf(val);
  }

  v8f o0 = {}, o1 = {}, o2 = {};
  float mrow[8], lrow[8];
#pragma unroll
  for (int r = 0; r < 8; ++r) { mrow[r] = -__builtin_inff(); lrow[r] = 0.0f; }

  const float scale = 0.14433756729740643f;  // 1/sqrt(48)
  const int ktmax = causal ? ((qt * 16 + 15) >> 5) : (Tk / 32 - 1);

  for (int kt = 0; kt <= ktmax; ++kt) {
    __syncthreads();
    const size_t kvbase = ((size_t)b * Tk + (size_t)kt * 32) * D_MODEL + h * HDIM;
    for (int idx = lane; idx < 32 * 64; idx += 32) {
      int kk = idx >> 6, kd = idx & 63;
      float val = (kd < HDIM) ? k[kvbase + (size_t)kk * D_MODEL + kd] : 0.0f;
      Ks[idx] = f2bf(val);
    }
    for (int idx = lane; idx < HDIM * 32; idx += 32) {
      int kk = idx & 31, kd = idx >> 5;
      Vts[kd * 40 + kk] = f2bf(v[kvbase + (size_t)kk * D_MODEL + kd]);
    }
    __syncthreads();

    // S = Q @ K^T for 32 keys (two 16-key fragments), K-dim 64 (padded)
    v8f s0 = {}, s1 = {};
#pragma unroll
    for (int ks = 0; ks < 64; ks += 32) {
      const unsigned short* ap = &Qs[ln16 * 64 + ks + hi * 16];
      v16bf af = load_frag(ap, ap + 8);
      const unsigned short* b0 = &Ks[ln16 * 64 + ks + hi * 16];
      v16bf bfr0 = load_frag(b0, b0 + 8);
      s0 = __builtin_amdgcn_wmma_f32_16x16x32_bf16(
          false, af, false, bfr0, (short)0, s0, false, false);
      const unsigned short* b1 = &Ks[(16 + ln16) * 64 + ks + hi * 16];
      v16bf bfr1 = load_frag(b1, b1 + 8);
      s1 = __builtin_amdgcn_wmma_f32_16x16x32_bf16(
          false, af, false, bfr1, (short)0, s1, false, false);
    }

    // online softmax update per accumulator row
#pragma unroll
    for (int r = 0; r < 8; ++r) {
      const int grow = qt * 16 + r + hi * 8;
      const int c0 = kt * 32 + ln16;
      const int c1 = c0 + 16;
      float x0 = s0[r] * scale, x1 = s1[r] * scale;
      if (causal) {
        if (c0 > grow) x0 = -__builtin_inff();
        if (c1 > grow) x1 = -__builtin_inff();
      }
      float mx = fmaxf(x0, x1);
      mx = fmaxf(mx, __shfl_xor(mx, 1, 32));
      mx = fmaxf(mx, __shfl_xor(mx, 2, 32));
      mx = fmaxf(mx, __shfl_xor(mx, 4, 32));
      mx = fmaxf(mx, __shfl_xor(mx, 8, 32));
      const float nm = fmaxf(mrow[r], mx);
      const float sc = (mrow[r] == -__builtin_inff()) ? 0.0f
                                                      : __expf(mrow[r] - nm);
      const float p0 = (x0 == -__builtin_inff()) ? 0.0f : __expf(x0 - nm);
      const float p1 = (x1 == -__builtin_inff()) ? 0.0f : __expf(x1 - nm);
      float rs = p0 + p1;
      rs += __shfl_xor(rs, 1, 32);
      rs += __shfl_xor(rs, 2, 32);
      rs += __shfl_xor(rs, 4, 32);
      rs += __shfl_xor(rs, 8, 32);
      lrow[r] = lrow[r] * sc + rs;
      mrow[r] = nm;
      o0[r] *= sc; o1[r] *= sc; o2[r] *= sc;
      Ps[(r + hi * 8) * 40 + ln16]      = f2bf(p0);
      Ps[(r + hi * 8) * 40 + 16 + ln16] = f2bf(p1);
    }
    __syncthreads();

    // O += P @ V   (K = 32 keys, three 16-wide HD tiles)
    const unsigned short* pp = &Ps[ln16 * 40 + hi * 16];
    v16bf pf = load_frag(pp, pp + 8);
    {
      const unsigned short* vp = &Vts[(0 * 16 + ln16) * 40 + hi * 16];
      v16bf vf = load_frag(vp, vp + 8);
      o0 = __builtin_amdgcn_wmma_f32_16x16x32_bf16(
          false, pf, false, vf, (short)0, o0, false, false);
    }
    {
      const unsigned short* vp = &Vts[(1 * 16 + ln16) * 40 + hi * 16];
      v16bf vf = load_frag(vp, vp + 8);
      o1 = __builtin_amdgcn_wmma_f32_16x16x32_bf16(
          false, pf, false, vf, (short)0, o1, false, false);
    }
    {
      const unsigned short* vp = &Vts[(2 * 16 + ln16) * 40 + hi * 16];
      v16bf vf = load_frag(vp, vp + 8);
      o2 = __builtin_amdgcn_wmma_f32_16x16x32_bf16(
          false, pf, false, vf, (short)0, o2, false, false);
    }
  }

#pragma unroll
  for (int r = 0; r < 8; ++r) {
    const int grow = qt * 16 + r + hi * 8;
    const float inv = 1.0f / lrow[r];
    const size_t base = ((size_t)b * T_DEC + grow) * D_MODEL + h * HDIM + ln16;
    o[base + 0]  = o0[r] * inv;
    o[base + 16] = o1[r] * inv;
    o[base + 32] = o2[r] * inv;
  }
}

// ---------------------------------------------------------------------------
// out = LayerNorm(x + res) * g + b   (res may be null). In-place safe.
// ---------------------------------------------------------------------------
__global__ __launch_bounds__(128) void addln_kernel(
    const float* __restrict__ x, const float* __restrict__ res,
    const float* __restrict__ g, const float* __restrict__ bta,
    float* __restrict__ out) {
  __shared__ float sb[128];
  const int row = blockIdx.x, tid = threadIdx.x;
  float vals[3];
  float s = 0.0f;
#pragma unroll
  for (int j = 0; j < 3; ++j) {
    int d = tid + j * 128;
    float vv = x[(size_t)row * D_MODEL + d];
    if (res) vv += res[(size_t)row * D_MODEL + d];
    vals[j] = vv;
    s += vv;
  }
  sb[tid] = s; __syncthreads();
  for (int st = 64; st > 0; st >>= 1) {
    if (tid < st) sb[tid] += sb[tid + st];
    __syncthreads();
  }
  const float mean = sb[0] / (float)D_MODEL;
  __syncthreads();
  float vsum = 0.0f;
#pragma unroll
  for (int j = 0; j < 3; ++j) { float dd = vals[j] - mean; vsum += dd * dd; }
  sb[tid] = vsum; __syncthreads();
  for (int st = 64; st > 0; st >>= 1) {
    if (tid < st) sb[tid] += sb[tid + st];
    __syncthreads();
  }
  const float rstd = rsqrtf(sb[0] / (float)D_MODEL + 1e-5f);
#pragma unroll
  for (int j = 0; j < 3; ++j) {
    int d = tid + j * 128;
    out[(size_t)row * D_MODEL + d] = (vals[j] - mean) * rstd * g[d] + bta[d];
  }
}

// ---------------------------------------------------------------------------
extern "C" void kernel_launch(void* const* d_in, const int* in_sizes, int n_in,
                              void* d_out, int out_size, void* d_ws, size_t ws_size,
                              hipStream_t stream) {
  (void)in_sizes; (void)n_in; (void)out_size; (void)ws_size;
  const float* enc    = (const float*)d_in[0];
  const float* emb    = (const float*)d_in[1];
  const float* sa_qw  = (const float*)d_in[2];
  const float* sa_qb  = (const float*)d_in[3];
  const float* sa_kw  = (const float*)d_in[4];
  const float* sa_kb  = (const float*)d_in[5];
  const float* sa_vw  = (const float*)d_in[6];
  const float* sa_vb  = (const float*)d_in[7];
  const float* sa_ow  = (const float*)d_in[8];
  const float* sa_ob  = (const float*)d_in[9];
  const float* ca_qw  = (const float*)d_in[10];
  const float* ca_qb  = (const float*)d_in[11];
  const float* ca_kw  = (const float*)d_in[12];
  const float* ca_kb  = (const float*)d_in[13];
  const float* ca_vw  = (const float*)d_in[14];
  const float* ca_vb  = (const float*)d_in[15];
  const float* ca_ow  = (const float*)d_in[16];
  const float* ca_ob  = (const float*)d_in[17];
  const float* ln1_g  = (const float*)d_in[18];
  const float* ln1_b  = (const float*)d_in[19];
  const float* ln2_g  = (const float*)d_in[20];
  const float* ln2_b  = (const float*)d_in[21];
  const float* ln3_g  = (const float*)d_in[22];
  const float* ln3_b  = (const float*)d_in[23];
  const float* ffn_w1 = (const float*)d_in[24];
  const float* ffn_b1 = (const float*)d_in[25];
  const float* ffn_w2 = (const float*)d_in[26];
  const float* ffn_b2 = (const float*)d_in[27];
  const float* on_g   = (const float*)d_in[28];
  const float* on_b   = (const float*)d_in[29];
  const float* out_w  = (const float*)d_in[30];
  const float* out_b  = (const float*)d_in[31];
  const int*   ids    = (const int*)d_in[32];
  float* logits = (float*)d_out;

  const int MQ = B_SZ * T_DEC;   // 2048
  const int MK = B_SZ * N_ENC;   // 4096

  float* x  = (float*)d_ws;
  float* qb = x  + (size_t)MQ * D_MODEL;
  float* kb = qb + (size_t)MQ * D_MODEL;
  float* vb = kb + (size_t)MK * D_MODEL;
  float* ao = vb + (size_t)MK * D_MODEL;
  float* h1 = ao + (size_t)MQ * D_MODEL;             // MQ x F_FFN scratch
  unsigned short* wb_cur = (unsigned short*)(h1 + (size_t)MQ * F_FFN);

  const dim3 blk256(256), blk128(128), blk32(32);

  // Convert fp32 weight [K][N] -> bf16 transposed [N][K] in workspace.
  auto conv = [&](const float* wsrc, int K, int N) -> const unsigned short* {
    unsigned short* dst = wb_cur;
    wb_cur += (size_t)K * N;
    int total = K * N;
    w2bf_kernel<<<dim3((total + 255) / 256), blk256, 0, stream>>>(wsrc, dst, K, N);
    return dst;
  };

  auto gemm0 = [&](const float* A, const unsigned short* Wb, const float* bias,
                   float* C, int M, int K, int Nc) {
    gemm_kernel<0><<<dim3(Nc / 64, M / 128), blk256, 0, stream>>>(A, Wb, bias, C, K, Nc);
  };
  auto gemm1 = [&](const float* A, const unsigned short* Wb, const float* bias,
                   float* C, int M, int K, int Nc) {
    gemm_kernel<1><<<dim3(Nc / 64, M / 128), blk256, 0, stream>>>(A, Wb, bias, C, K, Nc);
  };

  embed_kernel<<<dim3(MQ), blk128, 0, stream>>>(emb, ids, x);

  for (int i = 0; i < L_LAYERS; ++i) {
    const size_t wo  = (size_t)i * D_MODEL * D_MODEL;
    const size_t bo  = (size_t)i * D_MODEL;
    const size_t w1o = (size_t)i * D_MODEL * F_FFN;
    const size_t b1o = (size_t)i * F_FFN;
    const size_t w2o = (size_t)i * F_FFN * D_MODEL;

    // --- causal self-attention ---
    const unsigned short* qwb = conv(sa_qw + wo, D_MODEL, D_MODEL);
    const unsigned short* kwb = conv(sa_kw + wo, D_MODEL, D_MODEL);
    const unsigned short* vwb = conv(sa_vw + wo, D_MODEL, D_MODEL);
    const unsigned short* owb = conv(sa_ow + wo, D_MODEL, D_MODEL);
    gemm0(x, qwb, sa_qb + bo, qb, MQ, D_MODEL, D_MODEL);
    gemm0(x, kwb, sa_kb + bo, kb, MQ, D_MODEL, D_MODEL);
    gemm0(x, vwb, sa_vb + bo, vb, MQ, D_MODEL, D_MODEL);
    attn_kernel<<<dim3(T_DEC / 16, N_HEAD, B_SZ), blk32, 0, stream>>>(
        qb, kb, vb, ao, T_DEC, 1);
    gemm0(ao, owb, sa_ob + bo, h1, MQ, D_MODEL, D_MODEL);
    addln_kernel<<<dim3(MQ), blk128, 0, stream>>>(x, h1, ln1_g + bo, ln1_b + bo, x);

    // --- cross-attention ---
    const unsigned short* cqwb = conv(ca_qw + wo, D_MODEL, D_MODEL);
    const unsigned short* ckwb = conv(ca_kw + wo, D_MODEL, D_MODEL);
    const unsigned short* cvwb = conv(ca_vw + wo, D_MODEL, D_MODEL);
    const unsigned short* cowb = conv(ca_ow + wo, D_MODEL, D_MODEL);
    gemm0(x,   cqwb, ca_qb + bo, qb, MQ, D_MODEL, D_MODEL);
    gemm0(enc, ckwb, ca_kb + bo, kb, MK, D_MODEL, D_MODEL);
    gemm0(enc, cvwb, ca_vb + bo, vb, MK, D_MODEL, D_MODEL);
    attn_kernel<<<dim3(T_DEC / 16, N_HEAD, B_SZ), blk32, 0, stream>>>(
        qb, kb, vb, ao, N_ENC, 0);
    gemm0(ao, cowb, ca_ob + bo, h1, MQ, D_MODEL, D_MODEL);
    addln_kernel<<<dim3(MQ), blk128, 0, stream>>>(x, h1, ln2_g + bo, ln2_b + bo, x);

    // --- FFN ---
    const unsigned short* w1b = conv(ffn_w1 + w1o, D_MODEL, F_FFN);
    const unsigned short* w2b = conv(ffn_w2 + w2o, F_FFN, D_MODEL);
    gemm1(x,  w1b, ffn_b1 + b1o, h1, MQ, D_MODEL, F_FFN);
    gemm0(h1, w2b, ffn_b2 + bo,  ao, MQ, F_FFN, D_MODEL);
    addln_kernel<<<dim3(MQ), blk128, 0, stream>>>(x, ao, ln3_g + bo, ln3_b + bo, x);
  }

  addln_kernel<<<dim3(MQ), blk128, 0, stream>>>(x, nullptr, on_g, on_b, x);
  const unsigned short* outwb = conv(out_w, D_MODEL, V_VOCAB);
  gemm0(x, outwb, out_b, logits, MQ, D_MODEL, V_VOCAB);
}
